// MHSA_68298569941672
// MI455X (gfx1250) — compile-verified, matrix operand order checked by reading, not run on hardware
//
#include <hip/hip_runtime.h>
#include <hip/hip_bf16.h>

typedef __attribute__((ext_vector_type(16))) _Float16 v16h;
typedef __attribute__((ext_vector_type(8)))  _Float16 v8h;
typedef __attribute__((ext_vector_type(8)))  float    v8f;

constexpr int Bc = 16;     // batch
constexpr int Sc = 1024;   // sequence
constexpr int Hc = 16;     // heads
constexpr int DH = 64;     // head dim
constexpr int DM = 1024;   // model dim

// attention scale folded into Q at projection time: 1/sqrt(64) * log2(e)
constexpr float QSCALE = 0.125f * 1.4426950408889634f;

// ---------------------------------------------------------------------------
// WMMA helper: D = A(16x32 f16) * B(32x16 f16) + C(16x16 f32)
// ---------------------------------------------------------------------------
__device__ inline v8f wmma16(v16h a, v16h b, v8f c) {
  return __builtin_amdgcn_wmma_f32_16x16x32_f16(
      /*neg_a=*/false, a, /*neg_b=*/false, b,
      /*c_mod=*/(short)0, c, /*reuse_a=*/false, /*reuse_b=*/false);
}

// ---------------------------------------------------------------------------
// ds_swizzle group-of-32 encoding: offset[14:10]=xor, [9:5]=or, [4:0]=and
// new_lane = ((lane & and) | or) ^ xor
// ---------------------------------------------------------------------------
template <int XOR>
__device__ inline float swz(float v) {
  return __int_as_float(
      __builtin_amdgcn_ds_swizzle(__float_as_int(v), (XOR << 10) | 0x1f));
}
__device__ inline float rowmax16(float v) {
  v = fmaxf(v, swz<1>(v)); v = fmaxf(v, swz<2>(v));
  v = fmaxf(v, swz<4>(v)); v = fmaxf(v, swz<8>(v));
  return v;
}
// broadcast from lane 0 of each 16-lane half (BCASTX16: and=0x10, or=0, xor=0)
__device__ inline float bcast_half0(float v) {
  return __int_as_float(__builtin_amdgcn_ds_swizzle(__float_as_int(v), 0x10));
}

// ---------------------------------------------------------------------------
// Fragment loaders. CDNA5 16-bit A/B fragment, per lane (g = lane/16):
//   elem i (i<8) -> K = kbase + 8g + i ; elem 8+i -> K = kbase + 8g + 16 + i
// i.e. two contiguous runs of 8 at p and p+16.
// ---------------------------------------------------------------------------
__device__ inline v16h frag_from_f32(const float* __restrict__ p) {
  v16h a;
#pragma unroll
  for (int i = 0; i < 8; ++i) { a[i] = (_Float16)p[i]; a[8 + i] = (_Float16)p[16 + i]; }
  return a;
}
__device__ inline v16h frag_from_f16(const _Float16* __restrict__ p) {
  v8h lo = *(const v8h*)p;
  v8h hi = *(const v8h*)(p + 16);
  v16h a;
#pragma unroll
  for (int i = 0; i < 8; ++i) { a[i] = lo[i]; a[8 + i] = hi[i]; }
  return a;
}

// ===========================================================================
// Kernel 1: Q/K/V projections.  One wave = one (b,h) x 16-row s-tile.
// Bias rides in the WMMA C operand. Q is pre-scaled by QSCALE.
// Q,K: LDS-transposed then coalesced b128 stores, row-major [bh][s][dh].
// V  : stored transposed [bh][dh][s] with packed b128 stores.
// ===========================================================================
__global__ __launch_bounds__(256)
void qkv_kernel(const float* __restrict__ x,
                const float* __restrict__ Wq, const float* __restrict__ Wk,
                const float* __restrict__ Wv,
                const float* __restrict__ bq, const float* __restrict__ bk,
                const float* __restrict__ bv,
                _Float16* __restrict__ Qo, _Float16* __restrict__ Ko,
                _Float16* __restrict__ Vt) {
  __shared__ __align__(16) _Float16 Tr[8][16 * 72];  // per-wave 16x64 tile, pad 72

  const int lane = threadIdx.x & 31;
  const int wave = threadIdx.x >> 5;
  const int m = lane & 15, g = lane >> 4;
  const int blocksPerBH = Sc / 128;                 // 8
  const int bh = blockIdx.x / blocksPerBH;
  const int s0 = (blockIdx.x % blocksPerBH) * 128 + wave * 16;
  const int b = bh / Hc, h = bh % Hc;

  // A-fragments of X tile (rows s0..s0+15, cols h*64..h*64+63), f32 -> f16
  const float* xrow = x + ((size_t)b * Sc + (s0 + m)) * DM + h * DH;
  v16h a0 = frag_from_f32(xrow + 0  + g * 8);
  v16h a1 = frag_from_f32(xrow + 32 + g * 8);

  const float* Ws[3] = {Wq, Wk, Wv};
  const float* bs[3] = {bq, bk, bv};
  _Float16* Tw = &Tr[wave][0];

#pragma unroll
  for (int wsel = 0; wsel < 3; ++wsel) {
#pragma unroll
    for (int e0 = 0; e0 < DH; e0 += 16) {
      // B[k][n] = W[h][e0+n][k]  (y = x * W^T): lane n reads row e0+n of W
      const float* Wrow = Ws[wsel] + ((size_t)h * DH + (e0 + m)) * DH;
      const float bias = bs[wsel][h * DH + e0 + m];
      v8f acc;
#pragma unroll
      for (int i = 0; i < 8; ++i) acc[i] = bias;   // bias in C operand
      acc = wmma16(a0, frag_from_f32(Wrow + 0  + g * 8), acc);
      acc = wmma16(a1, frag_from_f32(Wrow + 32 + g * 8), acc);

      if (wsel == 2) {
        // V transposed: row e0+m, 8 consecutive s values -> one 16B store
        __align__(16) _Float16 tmp[8];
#pragma unroll
        for (int r = 0; r < 8; ++r) tmp[r] = (_Float16)acc[r];
        _Float16* dst = Vt + ((size_t)bh * DH + e0 + m) * Sc + s0 + g * 8;
        *(float4*)dst = *(const float4*)tmp;
      } else {
        const float sc = (wsel == 0) ? QSCALE : 1.0f;
#pragma unroll
        for (int r = 0; r < 8; ++r)
          Tw[(r + 8 * g) * 72 + e0 + m] = (_Float16)(acc[r] * sc);
      }
    }
    if (wsel < 2) {
      asm volatile("s_wait_dscnt 0x0" ::: "memory");
      // coalesced write-out: lane t handles row t/2, cols (t&1)*32..+31
      const int row = lane >> 1, cb = (lane & 1) * 32;
      const _Float16* srcp = Tw + row * 72 + cb;
      _Float16* dstp = (wsel == 0 ? Qo : Ko) +
                       ((size_t)bh * Sc + s0 + row) * DH + cb;
#pragma unroll
      for (int q = 0; q < 4; ++q)
        *(v8h*)(dstp + q * 8) = *(const v8h*)(srcp + q * 8);
      asm volatile("s_wait_dscnt 0x0" ::: "memory");  // before Tw reuse
    }
  }
}

// ===========================================================================
// Kernel 2: flash attention. One wave = 16 query rows of one (b,h).
// 32 keys per block; online softmax in exp2 domain (scale pre-folded in Q).
// Row-sum of P computed by an extra "ones-column" WMMA (l = P*1), which
// inherits the same alpha rescaling as O.
// ===========================================================================
__global__ __launch_bounds__(256)
void attn_kernel(const _Float16* __restrict__ Qi, const _Float16* __restrict__ Ki,
                 const _Float16* __restrict__ Vt, float* __restrict__ out) {
  __shared__ __align__(16) _Float16 Plds[8][16 * 32];  // per-wave P tile (1 KB)

  const int lane = threadIdx.x & 31;
  const int wave = threadIdx.x >> 5;
  const int m = lane & 15, g = lane >> 4;
  const int blocksPerBH = Sc / 128;
  const int bh = blockIdx.x / blocksPerBH;
  const int s0 = (blockIdx.x % blocksPerBH) * 128 + wave * 16;
  const int b = bh / Hc, h = bh % Hc;

  // Q A-fragments stay in registers for the whole key loop (already scaled)
  const _Float16* qrow = Qi + ((size_t)bh * Sc + s0 + m) * DH;
  v16h qa0 = frag_from_f16(qrow + 0  + g * 8);
  v16h qa1 = frag_from_f16(qrow + 32 + g * 8);

  // ones-column B fragment: B1[k][n] = (n==0); lanes 0/16 all-ones, else zero
  v16h ones;
  {
    const _Float16 o = (m == 0) ? (_Float16)1.0f : (_Float16)0.0f;
#pragma unroll
    for (int i = 0; i < 16; ++i) ones[i] = o;
  }

  float mrow[8];
  v8f oacc[4], oext;
  const v8f vzero = {0.f, 0.f, 0.f, 0.f, 0.f, 0.f, 0.f, 0.f};
#pragma unroll
  for (int r = 0; r < 8; ++r) mrow[r] = -1e30f;
#pragma unroll
  for (int et = 0; et < 4; ++et) oacc[et] = vzero;
  oext = vzero;

  _Float16* pw = &Plds[wave][0];

  for (int j = 0; j < Sc; j += 32) {
    // ---- scores for 32 keys: two 16x16 tiles, K-dim = d_head = 64 ----
    const _Float16* k_lo = Ki + ((size_t)bh * Sc + j + m) * DH;
    const _Float16* k_hi = k_lo + (size_t)16 * DH;
    v8f s_lo = vzero, s_hi = vzero;
    s_lo = wmma16(qa0, frag_from_f16(k_lo + 0  + g * 8), s_lo);
    s_lo = wmma16(qa1, frag_from_f16(k_lo + 32 + g * 8), s_lo);
    s_hi = wmma16(qa0, frag_from_f16(k_hi + 0  + g * 8), s_hi);
    s_hi = wmma16(qa1, frag_from_f16(k_hi + 32 + g * 8), s_hi);

    // ---- online softmax (row M = r + 8g lives across lanes of one half) ----
#pragma unroll
    for (int r = 0; r < 8; ++r) {
      const float slo = s_lo[r], shi = s_hi[r];  // already log2e-scaled
      const float mnew = fmaxf(mrow[r], rowmax16(fmaxf(slo, shi)));
      const float alpha = exp2f(mrow[r] - mnew);
      mrow[r] = mnew;
      const float plo = exp2f(slo - mnew);
      const float phi = exp2f(shi - mnew);
      oacc[0][r] *= alpha; oacc[1][r] *= alpha;
      oacc[2][r] *= alpha; oacc[3][r] *= alpha;
      oext[r] *= alpha;
      // P tile to LDS in plain row-major [M][k] (C-layout scatter)
      pw[(r + 8 * g) * 32 + m]      = (_Float16)plo;
      pw[(r + 8 * g) * 32 + 16 + m] = (_Float16)phi;
    }
    // wave-internal LDS store->load ordering (cross-lane, no block barrier)
    asm volatile("s_wait_dscnt 0x0" ::: "memory");

    // reload P as A-fragment (16x32, K = key-in-block)
    v16h pa = frag_from_f16(pw + m * 32 + g * 8);

    // ---- O += P * V ; l += P * 1 (ones column) ----
#pragma unroll
    for (int et = 0; et < 4; ++et) {
      const _Float16* vrow = Vt + ((size_t)bh * DH + et * 16 + m) * Sc + j + g * 8;
      oacc[et] = wmma16(pa, frag_from_f16(vrow), oacc[et]);
    }
    oext = wmma16(pa, ones, oext);
  }

  // ---- normalize and store: out[b][s][h*64 + e] ----
#pragma unroll
  for (int r = 0; r < 8; ++r) {
    const float l = bcast_half0(oext[r]);  // l lives in lane 0 of each half
    const float inv = 1.0f / l;
    float* orow = out + ((size_t)b * Sc + s0 + r + 8 * g) * DM + h * DH + m;
#pragma unroll
    for (int et = 0; et < 4; ++et) orow[et * 16] = oacc[et][r] * inv;
  }
}

// ===========================================================================
extern "C" void kernel_launch(void* const* d_in, const int* in_sizes, int n_in,
                              void* d_out, int out_size, void* d_ws, size_t ws_size,
                              hipStream_t stream) {
  (void)in_sizes; (void)n_in; (void)out_size; (void)ws_size;
  const float* x  = (const float*)d_in[0];
  const float* Wq = (const float*)d_in[1];
  const float* Wk = (const float*)d_in[2];
  const float* Wv = (const float*)d_in[3];
  const float* bq = (const float*)d_in[4];
  const float* bk = (const float*)d_in[5];
  const float* bv = (const float*)d_in[6];

  const size_t per = (size_t)Bc * Hc * Sc * DH;  // elements per Q/K/V plane
  _Float16* Q  = (_Float16*)d_ws;
  _Float16* K  = Q + per;
  _Float16* Vt = K + per;

  dim3 grid(Bc * Hc * (Sc / 128));  // 2048 blocks, 8 waves each
  qkv_kernel<<<grid, 256, 0, stream>>>(x, Wq, Wk, Wv, bq, bk, bv, Q, K, Vt);
  attn_kernel<<<grid, 256, 0, stream>>>(Q, K, Vt, (float*)d_out);
}
